// DeformConv_3934190043831
// MI455X (gfx1250) — matrix operand order repaired and checked by reading, not compiled, not run on hardware
//
#include <hip/hip_runtime.h>
#include <hip/hip_bf16.h>

typedef __attribute__((ext_vector_type(16))) _Float16 v16h;
typedef __attribute__((ext_vector_type(8)))  _Float16 v8h;
typedef __attribute__((ext_vector_type(8)))  float    v8f;
typedef __attribute__((ext_vector_type(4)))  int      v4i;

#define DCN_B   2
#define DCN_C   64
#define DCN_H   128
#define DCN_W   128
#define DCN_K   9
#define DCN_KD  576                     // C*K
#define DCN_M   (DCN_B*DCN_H*DCN_W)     // 32768
#define NSTEP   18                      // K / 32

#if defined(__has_builtin)
#if __has_builtin(__builtin_amdgcn_global_load_async_to_lds_b128)
#define DCN_ASYNC_LDS 1
#endif
#endif

// AS-qualified vector pointers for the async-to-LDS builtin
// (param0 prints as 'int4 __device__ *', i.e. addrspace(1) v4i*)
typedef __attribute__((address_space(1))) v4i* gv4i_p;
typedef __attribute__((address_space(3))) v4i* sv4i_p;

// ---------------------------------------------------------------------------
// Kernel 0: BN folding + weight conversion into WMMA-fragment-ordered fp16.
// Permuted K index q = k*64 + c  (q in [0,576)), so kernel1 can write the
// im2col matrix with contiguous 128B runs per tap.
// Fragment order: element e = ((s*4 + t)*32 + lane)*16 + i, where for step s,
// n-tile t:  n = t*16 + (lane&15),  q = s*32 + 16*(lane>>4) + i.
// ---------------------------------------------------------------------------
__global__ void dcn_prep(const float* __restrict__ w,
                         const float* __restrict__ bias,
                         const float* __restrict__ gamma,
                         const float* __restrict__ beta,
                         const float* __restrict__ rmean,
                         const float* __restrict__ rvar,
                         _Float16* __restrict__ BtF,
                         float* __restrict__ alpha,
                         float* __restrict__ cbeta) {
    int e = blockIdx.x * 256 + threadIdx.x;
    if (e < DCN_C * DCN_KD) {
        int i = e & 15;
        int l = (e >> 4) & 31;
        int f = e >> 9;                 // s*4 + t
        int t = f & 3;
        int s = f >> 2;
        int n = t * 16 + (l & 15);
        int q = s * 32 + ((l >> 4) << 4) + i;   // permuted K
        int c = q & 63;
        int k = q >> 6;
        BtF[e] = (_Float16)w[(size_t)n * DCN_KD + c * DCN_K + k];
    }
    if (e < DCN_C) {
        float inv = gamma[e] * rsqrtf(rvar[e] + 1e-5f);
        alpha[e] = inv;
        cbeta[e] = (bias[e] - rmean[e]) * inv + beta[e];
    }
}

// ---------------------------------------------------------------------------
// Kernel 1: offset conv (3x3, 64->27) + sigmoid mask + bilinear sampling.
// One thread per pixel; w_off (62KB) broadcast from LDS.
// Writes sampled fp16 [M][576] with column q = k*64 + c  (128B-contiguous per
// tap -> packed v8h/global_store_b128).
// ---------------------------------------------------------------------------
__global__ __launch_bounds__(256) void dcn_offset_sample(
    const float* __restrict__ x,
    const float* __restrict__ w_off,
    const float* __restrict__ b_off,
    _Float16* __restrict__ sampled) {

    extern __shared__ float smem[];
    float* sw = smem;            // [576][27] : sw[ct*27 + j] = w_off[j*576 + ct]
    float* sb = smem + 15552;    // [27]

    const int tid = threadIdx.x;
    for (int d = tid; d < 15552; d += 256) {
        int j  = d % 27;
        int ct = d / 27;
        sw[d] = w_off[(size_t)j * DCN_KD + ct];
    }
    if (tid < 27) sb[tid] = b_off[tid];
    __syncthreads();

    const int p  = blockIdx.x * 256 + tid;
    const int b  = p >> 14;
    const int hw = p & 16383;
    const int h  = hw >> 7;
    const int w  = hw & 127;

    float acc[27];
    #pragma unroll
    for (int j = 0; j < 27; ++j) acc[j] = sb[j];

    const float* xb = x + ((size_t)b << 20);
    for (int c = 0; c < DCN_C; ++c) {
        const float* xc = xb + ((size_t)c << 14);
        float xv[9];
        #pragma unroll
        for (int t = 0; t < 9; ++t) {
            int yy = h + t / 3 - 1;
            int xx = w + t % 3 - 1;
            bool ok = (yy >= 0) & (yy < DCN_H) & (xx >= 0) & (xx < DCN_W);
            xv[t] = ok ? xc[yy * DCN_W + xx] : 0.0f;
        }
        #pragma unroll
        for (int t = 0; t < 9; ++t) {
            const float* swp = &sw[(c * 9 + t) * 27];
            #pragma unroll
            for (int j = 0; j < 27; ++j) acc[j] += xv[t] * swp[j];
        }
    }

    _Float16* sp = sampled + (size_t)p * DCN_KD;
    #pragma unroll
    for (int k = 0; k < DCN_K; ++k) {
        float offy = acc[2 * k];
        float offx = acc[2 * k + 1];
        float msk  = 1.0f / (1.0f + __expf(-acc[18 + k]));
        float py = (float)(h - 1 + k / 3) + offy;
        float px = (float)(w - 1 + k % 3) + offx;
        float y0f = floorf(py), x0f = floorf(px);
        float ly = py - y0f, lx = px - x0f;
        int y0 = (int)y0f, x0 = (int)x0f;
        int y1 = y0 + 1,   x1 = x0 + 1;
        float vy0 = (y0 >= 0 && y0 < DCN_H) ? 1.0f : 0.0f;
        float vy1 = (y1 >= 0 && y1 < DCN_H) ? 1.0f : 0.0f;
        float vx0 = (x0 >= 0 && x0 < DCN_W) ? 1.0f : 0.0f;
        float vx1 = (x1 >= 0 && x1 < DCN_W) ? 1.0f : 0.0f;
        int yc0 = min(max(y0, 0), DCN_H - 1);
        int yc1 = min(max(y1, 0), DCN_H - 1);
        int xc0 = min(max(x0, 0), DCN_W - 1);
        int xc1 = min(max(x1, 0), DCN_W - 1);
        float w00 = (1.0f - ly) * (1.0f - lx) * msk * vy0 * vx0;
        float w01 = (1.0f - ly) * lx          * msk * vy0 * vx1;
        float w10 = ly          * (1.0f - lx) * msk * vy1 * vx0;
        float w11 = ly          * lx          * msk * vy1 * vx1;
        int i00 = yc0 * DCN_W + xc0, i01 = yc0 * DCN_W + xc1;
        int i10 = yc1 * DCN_W + xc0, i11 = yc1 * DCN_W + xc1;
        _Float16* spk = sp + k * 64;
        #pragma unroll
        for (int c0 = 0; c0 < DCN_C; c0 += 8) {
            v8h pk;
            #pragma unroll
            for (int u = 0; u < 8; ++u) {
                const float* xc = xb + ((size_t)(c0 + u) << 14);
                float s = w00 * xc[i00] + w01 * xc[i01]
                        + w10 * xc[i10] + w11 * xc[i11];
                pk[u] = (_Float16)s;
            }
            *(v8h*)(spk + c0) = pk;            // 16B aligned, b128 store
        }
    }
}

// ---------------------------------------------------------------------------
// Kernel 2: WMMA GEMM with B in LDS (fragment-ordered, async-copied),
// double-buffered A, fused BN+ReLU, coalesced NCHW store via padded LDS tile.
// Wave tile = 32(M) x 64(N): 8 accumulators, 8 WMMAs per k-step.
// Block = 8 waves = 256 M-rows; grid = 128 blocks.
// LDS: 73728B B-fragments (+stage reuse) | alpha 256B | cbeta 256B = 74240B.
// ---------------------------------------------------------------------------
__device__ __forceinline__ v16h load_afrag(const _Float16* pa) {
    v8h lo = *(const v8h*)(pa);
    v8h hi = *(const v8h*)(pa + 8);
    v16h r;
    #pragma unroll
    for (int i = 0; i < 8; ++i) { r[i] = lo[i]; r[i + 8] = hi[i]; }
    return r;
}

__global__ __launch_bounds__(256) void dcn_wmma_gemm(
    const _Float16* __restrict__ A,     // [32768][576] (permuted K)
    const _Float16* __restrict__ BtF,   // 36864 halves, fragment-ordered
    const float* __restrict__ alpha,
    const float* __restrict__ cbeta,
    float* __restrict__ out) {

    extern __shared__ char lds[];
    _Float16* ldsB = (_Float16*)lds;                  // 73728 B
    float*    aL   = (float*)(lds + 73728);           // 256 B
    float*    bL   = (float*)(lds + 73984);           // 256 B

    const int tid  = threadIdx.x;
    const int lane = tid & 31;
    const int wave = tid >> 5;
    const int lh   = lane & 15;
    const int hi   = lane >> 4;

    // ---- stage B fragments (72KB) into LDS, linear copy ----
#if DCN_ASYNC_LDS
    for (int i = tid; i < 4608; i += 256) {
        __builtin_amdgcn_global_load_async_to_lds_b128(
            (gv4i_p)(void*)(BtF + (size_t)i * 8),
            (sv4i_p)(void*)(lds + (size_t)i * 16),
            0, 0);
    }
    asm volatile("s_wait_asynccnt 0x0" ::: "memory");
#else
    for (int i = tid; i < 4608; i += 256) {
        ((uint4*)lds)[i] = ((const uint4*)BtF)[i];
    }
#endif
    if (tid < DCN_C) { aL[tid] = alpha[tid]; bL[tid] = cbeta[tid]; }
    __syncthreads();

    const int row0 = blockIdx.x * 256 + wave * 32;
    const _Float16* pa0 = A + (size_t)(row0 + lh) * DCN_KD + hi * 16;
    const _Float16* pa1 = pa0 + 16 * DCN_KD;

    v8f acc0[4] = {};
    v8f acc1[4] = {};

    v16h a0A = load_afrag(pa0);
    v16h a1A = load_afrag(pa1);

    for (int s = 0; s < NSTEP; s += 2) {
        // issue A loads for step s+1 before consuming step s
        v16h a0B = load_afrag(pa0 + (s + 1) * 32);
        v16h a1B = load_afrag(pa1 + (s + 1) * 32);
        __builtin_prefetch(pa0 + (s + 2) * 32, 0, 0);

        #pragma unroll
        for (int t = 0; t < 4; ++t) {
            const v16h bf = *(const v16h*)(ldsB + ((s * 4 + t) << 9) + (lane << 4));
            acc0[t] = __builtin_amdgcn_wmma_f32_16x16x32_f16(
                false, a0A, false, bf, (short)0, acc0[t], false, false);
            acc1[t] = __builtin_amdgcn_wmma_f32_16x16x32_f16(
                false, a1A, false, bf, (short)0, acc1[t], false, false);
        }

        if (s + 2 < NSTEP) {   // uniform branch
            a0A = load_afrag(pa0 + (s + 2) * 32);
            a1A = load_afrag(pa1 + (s + 2) * 32);
        }

        #pragma unroll
        for (int t = 0; t < 4; ++t) {
            const v16h bf = *(const v16h*)(ldsB + (((s + 1) * 4 + t) << 9) + (lane << 4));
            acc0[t] = __builtin_amdgcn_wmma_f32_16x16x32_f16(
                false, a0B, false, bf, (short)0, acc0[t], false, false);
            acc1[t] = __builtin_amdgcn_wmma_f32_16x16x32_f16(
                false, a1B, false, bf, (short)0, acc1[t], false, false);
        }
    }

    // ---- epilogue: BN+ReLU, transpose through padded LDS, coalesced store ----
    __syncthreads();                       // everyone done reading ldsB
    float* stage = (float*)lds;            // 256 x 65 floats = 66560 B (< 73728)

    const int mw = wave * 32 + hi * 8;
    #pragma unroll
    for (int t = 0; t < 4; ++t) {
        int n = t * 16 + lh;
        float ga = aL[n], gb = bL[n];
        #pragma unroll
        for (int r = 0; r < 8; ++r) {
            float v0 = acc0[t][r] * ga + gb;  v0 = v0 > 0.0f ? v0 : 0.0f;
            float v1 = acc1[t][r] * ga + gb;  v1 = v1 > 0.0f ? v1 : 0.0f;
            stage[(mw + r) * 65 + n]      = v0;
            stage[(mw + 16 + r) * 65 + n] = v1;
        }
    }
    __syncthreads();

    const int mblk = blockIdx.x << 8;
    for (int j = tid; j < 4096; j += 256) {
        int n  = j >> 6;
        int m4 = (j & 63) << 2;
        float4 v;
        v.x = stage[(m4 + 0) * 65 + n];
        v.y = stage[(m4 + 1) * 65 + n];
        v.z = stage[(m4 + 2) * 65 + n];
        v.w = stage[(m4 + 3) * 65 + n];
        int m  = mblk + m4;
        int bb = m >> 14;
        int hw = m & 16383;
        *(float4*)(out + (((size_t)(bb * DCN_C + n)) << 14) + hw) = v;
    }
}

// ---------------------------------------------------------------------------
extern "C" void kernel_launch(void* const* d_in, const int* in_sizes, int n_in,
                              void* d_out, int out_size, void* d_ws, size_t ws_size,
                              hipStream_t stream) {
    const float* x      = (const float*)d_in[0];
    const float* w_off  = (const float*)d_in[1];
    const float* b_off  = (const float*)d_in[2];
    const float* weight = (const float*)d_in[3];
    const float* bias   = (const float*)d_in[4];
    const float* gamma  = (const float*)d_in[5];
    const float* beta   = (const float*)d_in[6];
    const float* rmean  = (const float*)d_in[7];
    const float* rvar   = (const float*)d_in[8];
    float* out = (float*)d_out;

    char* ws = (char*)d_ws;
    _Float16* BtF     = (_Float16*)ws;               // 73,728 B (fragment order)
    float*    alpha   = (float*)(ws + 73728);        // 256 B
    float*    cbeta   = (float*)(ws + 73984);        // 256 B
    _Float16* sampled = (_Float16*)(ws + 74240);     // 37,748,736 B

    dcn_prep<<<144, 256, 0, stream>>>(weight, bias, gamma, beta, rmean, rvar,
                                      BtF, alpha, cbeta);

    dcn_offset_sample<<<DCN_M / 256, 256, 15579 * 4, stream>>>(x, w_off, b_off, sampled);

    dcn_wmma_gemm<<<DCN_M / 256, 256, 74240, stream>>>(sampled, BtF, alpha, cbeta, out);
}
